// ContxAggreBlock2D_28939489640667
// MI455X (gfx1250) — compile-verified, matrix operand order checked by reading, not run on hardware
//
#include <hip/hip_runtime.h>
#include <hip/hip_bf16.h>

typedef __bf16 bf16_t;
typedef __bf16 v16bf __attribute__((ext_vector_type(16)));
typedef float  v8f   __attribute__((ext_vector_type(8)));

constexpr int NPTS = 40960;
constexpr int GX = 160, GY = 160, GZ = 16;
constexpr int C    = 128;           // CIN == COUT
constexpr int T1   = 25;            // 5x5x1 taps
constexpr int TD   = 27;            // 3x3x3 taps
constexpr int OX = 80, OY = 80, OZ = 8;
constexpr int ND   = OX * OY * OZ;  // 51200 downsampled positions
constexpr float EPSV  = 1e-5f;
constexpr float SLOPE = 0.01f;

// ---------------- utility kernels ----------------

__global__ void k_init_i32(int* __restrict__ p, int n, int val) {
    int i = blockIdx.x * blockDim.x + threadIdx.x;
    if (i < n) p[i] = val;
}

__global__ void k_scatter_idx(const int* __restrict__ coords, int* __restrict__ idxmap, int n) {
    int i = blockIdx.x * blockDim.x + threadIdx.x;
    if (i >= n) return;
    int x = coords[i * 4 + 1], y = coords[i * 4 + 2], z = coords[i * 4 + 3];
    idxmap[(x * GY + y) * GZ + z] = i;
}

// rulebook for 5x5x1 submanifold conv: out[x,y,z] += in[x+kx-2, y+ky-2, z] * W[kx,ky]
__global__ void k_rule1(const int* __restrict__ coords, const int* __restrict__ idxmap,
                        int* __restrict__ nbr, int n) {
    int gid = blockIdx.x * blockDim.x + threadIdx.x;
    if (gid >= n * T1) return;
    int i = gid / T1, t = gid % T1;
    int kx = t / 5, ky = t % 5;
    int ix = coords[i * 4 + 1] + kx - 2;
    int iy = coords[i * 4 + 2] + ky - 2;
    int iz = coords[i * 4 + 3];
    int r = -1;
    if (ix >= 0 && ix < GX && iy >= 0 && iy < GY)
        r = idxmap[(ix * GY + iy) * GZ + iz];
    nbr[gid] = r;
}

// rulebook for strided 3x3x3 conv: down[ox,oy,oz] += dense[2ox+dx-1, 2oy+dy-1, 2oz+dz-1] * Wp[dx,dy,dz]
__global__ void k_ruleD(const int* __restrict__ idxmap, int* __restrict__ nbr) {
    int gid = blockIdx.x * blockDim.x + threadIdx.x;
    if (gid >= ND * TD) return;
    int o = gid / TD, t = gid % TD;
    int dx = t / 9, dy = (t % 9) / 3, dz = t % 3;
    int oz = o % OZ, oy = (o / OZ) % OY, ox = o / (OZ * OY);
    int ix = 2 * ox + dx - 1, iy = 2 * oy + dy - 1, iz = 2 * oz + dz - 1;
    int r = -1;
    if (ix >= 0 && ix < GX && iy >= 0 && iy < GY && iz >= 0 && iz < GZ)
        r = idxmap[(ix * GY + iy) * GZ + iz];
    nbr[gid] = r;
}

__global__ void k_cvt_bf16(const float* __restrict__ in, bf16_t* __restrict__ out, int n) {
    int i = blockIdx.x * blockDim.x + threadIdx.x;
    if (i < n) out[i] = (bf16_t)in[i];
}

// W[t][ci][co] (f32) -> Wt[t][co][ci] (bf16) so B fragments read contiguously
__global__ void k_w_transpose(const float* __restrict__ W, bf16_t* __restrict__ Wt, int taps) {
    int gid = blockIdx.x * blockDim.x + threadIdx.x;
    if (gid >= taps * C * C) return;
    int t  = gid / (C * C);
    int r  = gid % (C * C);
    int co = r / C, ci = r % C;
    Wt[gid] = (bf16_t)W[t * C * C + ci * C + co];
}

// ---------------- GroupNorm (+ optional LeakyReLU) ----------------
// one thread per (point, group-of-4-channels)
__global__ void k_groupnorm(const float* __restrict__ h,
                            const float* __restrict__ gamma, const float* __restrict__ beta,
                            bf16_t* __restrict__ ob, float* __restrict__ of,
                            int npts, int leaky) {
    int gid = blockIdx.x * blockDim.x + threadIdx.x;
    if (gid >= npts * 32) return;
    int p = gid >> 5, g = gid & 31;
    int base = p * C + g * 4;
    float4 v = *(const float4*)(h + base);
    float mu = 0.25f * (v.x + v.y + v.z + v.w);
    float dx = v.x - mu, dy = v.y - mu, dz = v.z - mu, dw = v.w - mu;
    float var = 0.25f * (dx * dx + dy * dy + dz * dz + dw * dw);
    float inv = rsqrtf(var + EPSV);
    float xs[4] = {dx, dy, dz, dw};
#pragma unroll
    for (int j = 0; j < 4; ++j) {
        int cch = g * 4 + j;
        float xn = xs[j] * inv * gamma[cch] + beta[cch];
        if (leaky) xn = xn >= 0.0f ? xn : SLOPE * xn;
        ob[base + j] = (bf16_t)xn;
        if (of) of[base + j] = xn;
    }
}

// ---------------- WMMA gather-GEMM sparse conv, LDS-staged ----------------
// Block = 128 threads = 4 waves. Block tile = 32 rows x 128 output channels.
// Per tap: gathered A rows (zero-masked) are staged once into 8KB of LDS, then
// each wave reads A fragments via ds_load_b128 (shared across its 2 n-tiles)
// and computes 2 m-tiles x 2 n-tiles = 4 WMMA per K-step (16 per tap).
// Dead-tap skip is a block-uniform __syncthreads_or vote, which also serves as
// the WAR barrier on the LDS tile, so EXEC is all-1s around every WMMA.
__global__ __launch_bounds__(128)
void k_spconv_wmma(const bf16_t* __restrict__ A,    // [rows][128] bf16
                   const bf16_t* __restrict__ Wt,   // [T][128(co)][128(ci)] bf16
                   const int*    __restrict__ nbr,  // [Mtot][T]
                   const float*  __restrict__ bias, // [128] or nullptr
                   float*        __restrict__ Out,  // [Mtot][128] f32
                   int Mtot, int T) {
    __shared__ bf16_t sA[32 * C];                // 8 KB A tile

    const int tid  = threadIdx.x;
    const int tile = blockIdx.x;                 // 32 rows per tile (Mtot % 32 == 0)
    const int lane = tid & 31;
    const int wave = tid >> 5;                   // 0..3
    const int half = lane >> 4;
    const int m    = lane & 15;
    const int nbase = wave * 32;

    // staging role: 4 threads per row, 32B each
    const int rl = tid >> 2;                     // row 0..31
    const int qt = tid & 3;                      // 64B quarter of the 256B row

    v8f acc00 = {}, acc01 = {}, acc10 = {}, acc11 = {};

    for (int t = 0; t < T; ++t) {
        int idx = nbr[(size_t)(tile * 32 + rl) * T + t];
        bool valid = idx >= 0;
        // block-uniform vote; also the WAR barrier protecting sA reuse
        if (__syncthreads_or((int)valid) == 0) continue;

        // prefetch next tap's 32KB weight tile (128 threads x 256B)
        if (t + 1 < T)
            __builtin_prefetch(Wt + ((size_t)(t + 1) * C) * C + (size_t)tid * 128, 0, 1);

        // stage gathered, zero-masked A rows into LDS
        {
            unsigned zm = valid ? 0xFFFFFFFFu : 0u;
            const uint4* src = (const uint4*)(A + (size_t)(valid ? idx : 0) * C + qt * 32);
            uint4 d0 = src[0], d1 = src[1];
            d0.x &= zm; d0.y &= zm; d0.z &= zm; d0.w &= zm;
            d1.x &= zm; d1.y &= zm; d1.z &= zm; d1.w &= zm;
            uint4* dst = (uint4*)(sA + rl * C + qt * 32);
            dst[0] = d0; dst[1] = d1;
        }
        __syncthreads();

        const bf16_t* a0 = sA + m * C;           // m-tile 0: rows 0..15
        const bf16_t* a1 = sA + (16 + m) * C;    // m-tile 1: rows 16..31
        const bf16_t* b0 = Wt + ((size_t)t * C + (nbase + m)) * C;        // n-tile 0
        const bf16_t* b1 = Wt + ((size_t)t * C + (nbase + 16 + m)) * C;   // n-tile 1
#pragma unroll
        for (int ks = 0; ks < 4; ++ks) {
            const int kc = ks * 32 + half * 8;   // 16-bit operand: runs [8h..] and [16+8h..]
            union { v16bf v; uint4 q[2]; } au0, au1, bu0, bu1;
            au0.q[0] = *(const uint4*)(a0 + kc);
            au0.q[1] = *(const uint4*)(a0 + kc + 16);
            au1.q[0] = *(const uint4*)(a1 + kc);
            au1.q[1] = *(const uint4*)(a1 + kc + 16);
            bu0.q[0] = *(const uint4*)(b0 + kc);
            bu0.q[1] = *(const uint4*)(b0 + kc + 16);
            bu1.q[0] = *(const uint4*)(b1 + kc);
            bu1.q[1] = *(const uint4*)(b1 + kc + 16);
            acc00 = __builtin_amdgcn_wmma_f32_16x16x32_bf16(false, au0.v, false, bu0.v,
                                                            (short)0, acc00, false, false);
            acc01 = __builtin_amdgcn_wmma_f32_16x16x32_bf16(false, au0.v, false, bu1.v,
                                                            (short)0, acc01, false, false);
            acc10 = __builtin_amdgcn_wmma_f32_16x16x32_bf16(false, au1.v, false, bu0.v,
                                                            (short)0, acc10, false, false);
            acc11 = __builtin_amdgcn_wmma_f32_16x16x32_bf16(false, au1.v, false, bu1.v,
                                                            (short)0, acc11, false, false);
        }
    }

    // C/D layout: VGPR v holds M = v + 8*half, N = lane & 15
    int c0 = nbase + m, c1 = nbase + 16 + m;
    float bb0 = bias ? bias[c0] : 0.0f;
    float bb1 = bias ? bias[c1] : 0.0f;
#pragma unroll
    for (int v = 0; v < 8; ++v) {
        size_t r0 = (size_t)(tile * 32 + v + half * 8);       // m-tile 0
        size_t r1 = r0 + 16;                                  // m-tile 1
        Out[r0 * C + c0] = acc00[v] + bb0;
        Out[r0 * C + c1] = acc01[v] + bb1;
        Out[r1 * C + c0] = acc10[v] + bb0;
        Out[r1 * C + c1] = acc11[v] + bb1;
    }
}

// ---------------- launcher ----------------

extern "C" void kernel_launch(void* const* d_in, const int* in_sizes, int n_in,
                              void* d_out, int out_size, void* d_ws, size_t ws_size,
                              hipStream_t stream) {
    const float* feats  = (const float*)d_in[0];
    const int*   coords = (const int*)d_in[1];
    const float* W1 = (const float*)d_in[2];
    const float* b1 = (const float*)d_in[3];
    const float* g1 = (const float*)d_in[4];
    const float* be1 = (const float*)d_in[5];
    const float* W2 = (const float*)d_in[6];
    const float* b2 = (const float*)d_in[7];
    const float* g2 = (const float*)d_in[8];
    const float* be2 = (const float*)d_in[9];
    const float* Wp = (const float*)d_in[10];

    float* down_out  = (float*)d_out;                       // (1,80,80,8,128)
    float* feats_out = (float*)d_out + (size_t)ND * C;      // (N,128)

    // workspace carve (256B aligned)
    char* ws = (char*)d_ws;
    size_t off = 0;
    auto carve = [&](size_t bytes) {
        off = (off + 255) & ~(size_t)255;
        void* p = ws + off;
        off += bytes;
        return p;
    };
    int*    idxmap = (int*)carve((size_t)GX * GY * GZ * 4);
    int*    nbr1   = (int*)carve((size_t)NPTS * T1 * 4);
    int*    nbrD   = (int*)carve((size_t)ND * TD * 4);
    bf16_t* featsb = (bf16_t*)carve((size_t)NPTS * C * 2);
    bf16_t* W1t    = (bf16_t*)carve((size_t)T1 * C * C * 2);
    bf16_t* W2t    = (bf16_t*)carve((size_t)T1 * C * C * 2);
    bf16_t* Wpt    = (bf16_t*)carve((size_t)TD * C * C * 2);
    float*  h1     = (float*)carve((size_t)NPTS * C * 4);
    bf16_t* hb     = (bf16_t*)carve((size_t)NPTS * C * 2);
    float*  h2     = (float*)carve((size_t)NPTS * C * 4);
    bf16_t* ofb    = (bf16_t*)carve((size_t)NPTS * C * 2);
    (void)ws_size;

    const int TB = 256;
    auto nb = [](int n, int tb) { return (n + tb - 1) / tb; };

    // 1) index map + rulebooks
    k_init_i32<<<nb(GX * GY * GZ, TB), TB, 0, stream>>>(idxmap, GX * GY * GZ, -1);
    k_scatter_idx<<<nb(NPTS, TB), TB, 0, stream>>>(coords, idxmap, NPTS);
    k_rule1<<<nb(NPTS * T1, TB), TB, 0, stream>>>(coords, idxmap, nbr1, NPTS);
    k_ruleD<<<nb(ND * TD, TB), TB, 0, stream>>>(idxmap, nbrD);

    // 2) bf16 conversions
    k_cvt_bf16<<<nb(NPTS * C, TB), TB, 0, stream>>>(feats, featsb, NPTS * C);
    k_w_transpose<<<nb(T1 * C * C, TB), TB, 0, stream>>>(W1, W1t, T1);
    k_w_transpose<<<nb(T1 * C * C, TB), TB, 0, stream>>>(W2, W2t, T1);
    k_w_transpose<<<nb(TD * C * C, TB), TB, 0, stream>>>(Wp, Wpt, TD);

    // 3) conv1 -> GN+LeakyReLU -> conv2 -> GN (out_feats)
    k_spconv_wmma<<<NPTS / 32, 128, 0, stream>>>(featsb, W1t, nbr1, b1, h1, NPTS, T1);
    k_groupnorm<<<nb(NPTS * 32, TB), TB, 0, stream>>>(h1, g1, be1, hb, nullptr, NPTS, 1);
    k_spconv_wmma<<<NPTS / 32, 128, 0, stream>>>(hb, W2t, nbr1, b2, h2, NPTS, T1);
    k_groupnorm<<<nb(NPTS * 32, TB), TB, 0, stream>>>(h2, g2, be2, ofb, feats_out, NPTS, 0);

    // 4) strided 3x3x3 downsample conv (mask is implicit: no taps -> zero accumulator)
    k_spconv_wmma<<<ND / 32, 128, 0, stream>>>(ofb, Wpt, nbrD, nullptr, down_out, ND, TD);
}
